// GroupedQueryAttention_52432960749775
// MI455X (gfx1250) — matrix operand order repaired and checked by reading, not compiled, not run on hardware
//
#include <hip/hip_runtime.h>

// ---------------------------------------------------------------------------
// GQA forward for MI455X (gfx1250), wave32 + WMMA f16->f32.
// B=2, S=2048, H=32, HKV=1, DH=128.  ~146 GFLOP, compute-bound -> WMMA.
// Round 2: pipelined LDS fragment loads, 8-wave blocks, global prefetch.
// ---------------------------------------------------------------------------

typedef _Float16 half8  __attribute__((ext_vector_type(8)));
typedef _Float16 half16 __attribute__((ext_vector_type(16)));
typedef float    float8 __attribute__((ext_vector_type(8)));

union Frag16 {
    half16 v;
    half8  h[2];
};

static __device__ __forceinline__
float8 wmma_f16(const Frag16& a, const Frag16& b, float8 c) {
    // D = A(16x32 f16) * B(32x16 f16) + C(16x16 f32)
    return __builtin_amdgcn_wmma_f32_16x16x32_f16(
        false, a.v, false, b.v, (short)0, c, false, false);
}

static __device__ __forceinline__ float rowmax16(float v) {
    v = fmaxf(v, __shfl_xor(v, 1));
    v = fmaxf(v, __shfl_xor(v, 2));
    v = fmaxf(v, __shfl_xor(v, 4));
    v = fmaxf(v, __shfl_xor(v, 8));
    return v;
}
static __device__ __forceinline__ float rowsum16(float v) {
    v += __shfl_xor(v, 1);
    v += __shfl_xor(v, 2);
    v += __shfl_xor(v, 4);
    v += __shfl_xor(v, 8);
    return v;
}

// --------------------------- conversion kernels ----------------------------

__global__ void cvt_f32_to_f16(const float* __restrict__ src,
                               _Float16* __restrict__ dst, int n) {
    int i = blockIdx.x * blockDim.x + threadIdx.x;
    if (i < n) dst[i] = (_Float16)src[i];
}

// src: f32 [K][N] row-major  ->  dst: f16 [N][K] row-major (transposed)
__global__ void transpose_f32_to_f16(const float* __restrict__ src,
                                     _Float16* __restrict__ dst, int K, int N) {
    int idx = blockIdx.x * blockDim.x + threadIdx.x;
    if (idx < K * N) {
        int k = idx / N;
        int n = idx - k * N;
        dst[(size_t)n * K + k] = (_Float16)src[idx];
    }
}

// ------------------------------ WMMA GEMM ----------------------------------
// C[M,N] = A[M,K] (f16, row-major) * BT[N,K]^T (f16) + bias[N]
// Block: 128 threads = 4 waves; wave w -> rows [by*64+w*16, +16), cols [bx*64,+64)
template <bool OUT_F16>
__global__ __launch_bounds__(128)
void gemm_wmma(const _Float16* __restrict__ A,
               const _Float16* __restrict__ BT,
               const float* __restrict__ bias,
               void* __restrict__ outp,
               int M, int N, int K) {
    const int lane   = threadIdx.x & 31;
    const int wave   = threadIdx.x >> 5;
    const int laneHi = lane >> 4;   // 0 or 1
    const int laneLo = lane & 15;
    const int m0 = blockIdx.y * 64 + wave * 16;
    const int n0 = blockIdx.x * 64;

    float8 acc[4] = {};

    for (int kb = 0; kb < K; kb += 32) {
        // A fragment: row m0+laneLo; halves 0..7 at kb+8*laneHi, 8..15 at +16
        Frag16 a;
        const _Float16* arow = A + (size_t)(m0 + laneLo) * K + kb + laneHi * 8;
        a.h[0] = *(const half8*)(arow);
        a.h[1] = *(const half8*)(arow + 16);
        // batch all 4 B-fragment loads before the WMMA chain
        Frag16 b[4];
#pragma unroll
        for (int t = 0; t < 4; ++t) {
            const _Float16* brow =
                BT + (size_t)(n0 + t * 16 + laneLo) * K + kb + laneHi * 8;
            b[t].h[0] = *(const half8*)(brow);
            b[t].h[1] = *(const half8*)(brow + 16);
        }
#pragma unroll
        for (int t = 0; t < 4; ++t)
            acc[t] = wmma_f16(a, b[t], acc[t]);
    }

#pragma unroll
    for (int t = 0; t < 4; ++t) {
#pragma unroll
        for (int r = 0; r < 8; ++r) {
            int row = m0 + r + laneHi * 8;
            int col = n0 + t * 16 + laneLo;
            float v = acc[t][r] + bias[col];
            if (OUT_F16)
                ((_Float16*)outp)[(size_t)row * N + col] = (_Float16)v;
            else
                ((float*)outp)[(size_t)row * N + col] = v;
        }
    }
}

// --------------------------- flash attention -------------------------------
// Q: [B][H*S][128] f16 (reinterpreted Qproj), K/V: [B][S][128] f16,
// O: [B][S][H*128] f16.
// Block: 256 threads = 8 waves x 16 query rows; 32-key tiles staged in LDS
// and shared by all 8 waves; next tile prefetched into cache during compute.
__global__ __launch_bounds__(256)
void attn_wmma(const _Float16* __restrict__ Q,
               const _Float16* __restrict__ Kp,
               const _Float16* __restrict__ Vp,
               _Float16* __restrict__ O) {
    constexpr int S = 2048, DH = 128, H = 32;

    __shared__ __attribute__((aligned(16))) _Float16 Kt[32 * 128];  // [key][d]
    __shared__ __attribute__((aligned(16))) _Float16 Vt[128 * 32];  // [d][key]
    __shared__ __attribute__((aligned(16))) _Float16 Pbuf[8][16 * 32];

    const int tid    = threadIdx.x;
    const int lane   = tid & 31;
    const int wave   = tid >> 5;
    const int laneHi = lane >> 4;
    const int laneLo = lane & 15;
    const int b      = blockIdx.z;
    const int h      = blockIdx.y;
    const int sBase  = blockIdx.x * 128 + wave * 16;

    // Raw-reshape semantics: head h = rows [h*S, (h+1)*S) of Qproj flat [H*S,128]
    const _Float16* Qh = Q + ((size_t)b * H * S + (size_t)h * S + sBase) * DH;
    const _Float16* Kb = Kp + (size_t)b * S * DH;
    const _Float16* Vb = Vp + (size_t)b * S * DH;

    // Preload Q A-fragments for the 4 K-steps (DH=128 = 4*32)
    Frag16 qf[4];
#pragma unroll
    for (int ks = 0; ks < 4; ++ks) {
        const _Float16* qrow = Qh + (size_t)laneLo * DH + ks * 32 + laneHi * 8;
        qf[ks].h[0] = *(const half8*)(qrow);
        qf[ks].h[1] = *(const half8*)(qrow + 16);
    }

    float8 accO[8] = {};
    float  m[8], l[8];
#pragma unroll
    for (int r = 0; r < 8; ++r) { m[r] = -1e30f; l[r] = 0.0f; }

    const float scale = 0.08838834764831845f;  // 1/sqrt(128)

    for (int kt = 0; kt < S; kt += 32) {
        __syncthreads();  // prior iteration done reading tiles
        {
            // K tile row-major, 64-bit chunks (coalesced)
            const unsigned long long* src =
                (const unsigned long long*)(Kb + (size_t)kt * DH);
            unsigned long long* dst = (unsigned long long*)Kt;
            for (int i = tid; i < 32 * 128 / 4; i += 256) dst[i] = src[i];
            // V tile transposed: Vt[d][key]
            for (int i = tid; i < 32 * 128; i += 256) {
                int d = i & 127, key = i >> 7;
                Vt[d * 32 + key] = Vb[(size_t)(kt + key) * DH + d];
            }
        }
        __syncthreads();

        // prefetch next tile's cachelines (global_prefetch_b8) during compute
        if (kt + 32 < S && tid < 128) {
            __builtin_prefetch(Kb + (size_t)(kt + 32) * DH + tid * 32, 0, 1);
            __builtin_prefetch(Vb + (size_t)(kt + 32) * DH + tid * 32, 0, 1);
        }

        // ---- scores: preload all 8 K-tile B-fragments, then 8 WMMAs ----
        Frag16 bk[8];  // [2*ks + tile]
#pragma unroll
        for (int ks = 0; ks < 4; ++ks) {
            const _Float16* k0 = &Kt[laneLo * DH + ks * 32 + laneHi * 8];
            bk[2 * ks].h[0] = *(const half8*)(k0);
            bk[2 * ks].h[1] = *(const half8*)(k0 + 16);
            const _Float16* k1 = &Kt[(16 + laneLo) * DH + ks * 32 + laneHi * 8];
            bk[2 * ks + 1].h[0] = *(const half8*)(k1);
            bk[2 * ks + 1].h[1] = *(const half8*)(k1 + 16);
        }
        float8 s0 = {}, s1 = {};
#pragma unroll
        for (int ks = 0; ks < 4; ++ks) {
            s0 = wmma_f16(qf[ks], bk[2 * ks], s0);
            s1 = wmma_f16(qf[ks], bk[2 * ks + 1], s1);
        }

        // ---- online softmax per row (row r+8*laneHi, 16 lanes hold cols) ----
#pragma unroll
        for (int r = 0; r < 8; ++r) {
            float a0 = s0[r] * scale;
            float a1 = s1[r] * scale;
            float mx = rowmax16(fmaxf(a0, a1));
            float mNew  = fmaxf(m[r], mx);
            float alpha = __expf(m[r] - mNew);
            float p0 = __expf(a0 - mNew);
            float p1 = __expf(a1 - mNew);
            float rs = rowsum16(p0 + p1);
            l[r] = l[r] * alpha + rs;
            m[r] = mNew;
#pragma unroll
            for (int c = 0; c < 8; ++c) accO[c][r] *= alpha;
            // stash P (C-layout -> LDS, re-read as A-layout)
            int row = r + laneHi * 8;
            Pbuf[wave][row * 32 + laneLo]      = (_Float16)p0;
            Pbuf[wave][row * 32 + 16 + laneLo] = (_Float16)p1;
        }

        // ---- O += P(16x32) @ V(32x128): V-fragments in groups of 4 ----
        Frag16 pf;
        const _Float16* pr = &Pbuf[wave][laneLo * 32 + laneHi * 8];
        pf.h[0] = *(const half8*)(pr);
        pf.h[1] = *(const half8*)(pr + 16);
#pragma unroll
        for (int g = 0; g < 2; ++g) {
            Frag16 vf[4];
#pragma unroll
            for (int j = 0; j < 4; ++j) {
                int c = g * 4 + j;
                const _Float16* vr = &Vt[(c * 16 + laneLo) * 32 + laneHi * 8];
                vf[j].h[0] = *(const half8*)(vr);
                vf[j].h[1] = *(const half8*)(vr + 16);
            }
#pragma unroll
            for (int j = 0; j < 4; ++j)
                accO[g * 4 + j] = wmma_f16(pf, vf[j], accO[g * 4 + j]);
        }
    }

    // ---- write O (transposed head layout: [b][s][h*128+d]) ----
#pragma unroll
    for (int c = 0; c < 8; ++c) {
#pragma unroll
        for (int r = 0; r < 8; ++r) {
            int s = sBase + r + laneHi * 8;
            int d = c * 16 + laneLo;
            float v = accO[c][r] / l[r];
            O[((size_t)b * S + s) * (H * DH) + h * DH + d] = (_Float16)v;
        }
    }
}

// ------------------------------- launcher ----------------------------------

extern "C" void kernel_launch(void* const* d_in, const int* in_sizes, int n_in,
                              void* d_out, int out_size, void* d_ws, size_t ws_size,
                              hipStream_t stream) {
    (void)in_sizes; (void)n_in; (void)out_size; (void)ws_size;

    constexpr int B = 2, S = 2048, H = 32, DH = 128;
    constexpr int BS = B * S;        // 4096
    constexpr int HD = H * DH;       // 4096

    const float* query = (const float*)d_in[0];
    const float* key   = (const float*)d_in[1];
    const float* value = (const float*)d_in[2];
    const float* W_q   = (const float*)d_in[3];
    const float* b_q   = (const float*)d_in[4];
    const float* W_k   = (const float*)d_in[5];
    const float* b_k   = (const float*)d_in[6];
    const float* W_v   = (const float*)d_in[7];
    const float* b_v   = (const float*)d_in[8];
    const float* W_o   = (const float*)d_in[9];
    const float* b_o   = (const float*)d_in[10];
    float* out = (float*)d_out;

    // workspace layout (256B aligned slices)
    size_t off = 0;
    auto alloc = [&](size_t bytes) {
        void* p = (char*)d_ws + off;
        off += (bytes + 255) & ~(size_t)255;
        return p;
    };
    _Float16* q16  = (_Float16*)alloc((size_t)BS * DH * 2);
    _Float16* k16  = (_Float16*)alloc((size_t)BS * DH * 2);
    _Float16* v16  = (_Float16*)alloc((size_t)BS * DH * 2);
    _Float16* WqT  = (_Float16*)alloc((size_t)HD * DH * 2);   // [4096][128]
    _Float16* WkT  = (_Float16*)alloc((size_t)DH * DH * 2);   // [128][128]
    _Float16* WvT  = (_Float16*)alloc((size_t)DH * DH * 2);
    _Float16* WoT  = (_Float16*)alloc((size_t)DH * HD * 2);   // [128][4096]
    _Float16* Qp   = (_Float16*)alloc((size_t)BS * HD * 2);   // [4096][4096]
    _Float16* Kp   = (_Float16*)alloc((size_t)BS * DH * 2);
    _Float16* Vp   = (_Float16*)alloc((size_t)BS * DH * 2);
    _Float16* Oc   = (_Float16*)alloc((size_t)BS * HD * 2);   // [4096][4096]

    // 1) convert activations to f16
    {
        int n = BS * DH;
        int g = (n + 255) / 256;
        cvt_f32_to_f16<<<g, 256, 0, stream>>>(query, q16, n);
        cvt_f32_to_f16<<<g, 256, 0, stream>>>(key,   k16, n);
        cvt_f32_to_f16<<<g, 256, 0, stream>>>(value, v16, n);
    }
    // 2) transpose + convert weights: [K][N] f32 -> [N][K] f16
    {
        int n1 = DH * HD;   // W_q: K=128, N=4096
        transpose_f32_to_f16<<<(n1 + 255) / 256, 256, 0, stream>>>(W_q, WqT, DH, HD);
        int n2 = DH * DH;
        transpose_f32_to_f16<<<(n2 + 255) / 256, 256, 0, stream>>>(W_k, WkT, DH, DH);
        transpose_f32_to_f16<<<(n2 + 255) / 256, 256, 0, stream>>>(W_v, WvT, DH, DH);
        int n3 = HD * DH;   // W_o: K=4096, N=128
        transpose_f32_to_f16<<<(n3 + 255) / 256, 256, 0, stream>>>(W_o, WoT, HD, DH);
    }
    // 3) projections (f16 out)
    gemm_wmma<true><<<dim3(HD / 64, BS / 64), 128, 0, stream>>>(
        q16, WqT, b_q, (void*)Qp, BS, HD, DH);
    gemm_wmma<true><<<dim3(DH / 64, BS / 64), 128, 0, stream>>>(
        k16, WkT, b_k, (void*)Kp, BS, DH, DH);
    gemm_wmma<true><<<dim3(DH / 64, BS / 64), 128, 0, stream>>>(
        v16, WvT, b_v, (void*)Vp, BS, DH, DH);
    // 4) flash attention (scores never hit memory)
    attn_wmma<<<dim3(S / 128, H, B), 256, 0, stream>>>(Qp, Kp, Vp, Oc);
    // 5) output projection (f32 out + bias)
    gemm_wmma<false><<<dim3(DH / 64, BS / 64), 128, 0, stream>>>(
        Oc, WoT, b_o, (void*)out, BS, DH, HD);
}
